// Block_90890097918445
// MI455X (gfx1250) — compile-verified, matrix operand order checked by reading, not run on hardware
//
#include <hip/hip_runtime.h>
#include <math.h>

// ---------------- types ----------------
typedef __bf16 v16bf __attribute__((ext_vector_type(16)));
typedef float  v8f   __attribute__((ext_vector_type(8)));

union Frag {
  v16bf v;
  unsigned int u[8];
  unsigned short h[16];
};

__device__ __forceinline__ unsigned short f2bf(float f) {
  unsigned int u = __builtin_bit_cast(unsigned int, f);
  u += 0x7FFFu + ((u >> 16) & 1u);          // round-to-nearest-even
  return (unsigned short)(u >> 16);
}

// element-pair K index for 16-bit WMMA A/B fragments (ISA 7.12.2):
// VGPR v<4 holds k={2v,2v+1}+8*half ; v>=4 holds k={16+2(v-4),...}+8*half
__device__ __forceinline__ int pairK(int v, int half) {
  return ((v < 4) ? (2 * v) : (16 + 2 * (v - 4))) + 8 * half;
}

__device__ __forceinline__ v8f wmma_bf16(const Frag& a, const Frag& b, v8f c) {
  return __builtin_amdgcn_wmma_f32_16x16x32_bf16(
      false, a.v, false, b.v, (short)0, c, false, false);
}

// ---- CDNA5 async global->LDS DMA (ASYNCcnt-tracked, no VGPR data) ----
__device__ __forceinline__ void async_ld_b128(unsigned lds_byte_addr, const void* gptr) {
  asm volatile("global_load_async_to_lds_b128 %0, %1, off"
               :: "v"(lds_byte_addr), "v"(gptr) : "memory");
}
__device__ __forceinline__ void wait_async0() {
  asm volatile("s_wait_asynccnt 0x0" ::: "memory");
}
// generic LDS pointer -> 32-bit LDS byte offset (aperture keeps offset in addr[31:0])
__device__ __forceinline__ unsigned lds_addr(const void* p) {
  return (unsigned)(unsigned long long)p;
}

// ---------------- fp32 -> bf16 convert ----------------
__global__ __launch_bounds__(256) void cvt_bf16_kernel(
    const float* __restrict__ in, unsigned short* __restrict__ out, int n) {
  int i = blockIdx.x * 1024 + threadIdx.x;
#pragma unroll
  for (int k = 0; k < 4; ++k, i += 256)
    if (i < n) out[i] = f2bf(in[i]);
}

// ---------------- LayerNorm (per token, E=1024) ----------------
__global__ __launch_bounds__(256) void ln_bf16_kernel(
    const float* __restrict__ x, const float* __restrict__ g,
    const float* __restrict__ b, unsigned short* __restrict__ out) {
  const int E = 1024;
  int tok = blockIdx.x;
  const float* xr = x + (size_t)tok * E;
  float v0[4], s = 0.f, s2 = 0.f;
#pragma unroll
  for (int i = 0; i < 4; ++i) {
    float t = xr[threadIdx.x + i * 256];
    v0[i] = t; s += t; s2 += t * t;
  }
#pragma unroll
  for (int off = 16; off; off >>= 1) {
    s  += __shfl_xor(s,  off, 32);
    s2 += __shfl_xor(s2, off, 32);
  }
  __shared__ float ws[8], ws2[8];
  if ((threadIdx.x & 31) == 0) { ws[threadIdx.x >> 5] = s; ws2[threadIdx.x >> 5] = s2; }
  __syncthreads();
  float S = 0.f, S2 = 0.f;
#pragma unroll
  for (int i = 0; i < 8; ++i) { S += ws[i]; S2 += ws2[i]; }
  float mu = S * (1.0f / E);
  float var = S2 * (1.0f / E) - mu * mu;
  float rinv = rsqrtf(var + 1e-5f);
#pragma unroll
  for (int i = 0; i < 4; ++i) {
    int c = threadIdx.x + i * 256;
    out[(size_t)tok * E + c] = f2bf((v0[i] - mu) * rinv * g[c] + b[c]);
  }
}

// ---------------- bf16 GEMM: C = A[M,K] @ B[K,N] (+bias,+relu,+res) ----------------
#define GBM 128
#define GBN 64
#define GBK 32
#define GF_OUT_BF16 1
#define GF_RELU     2

template <int NDIM, int KDIM>
__global__ __launch_bounds__(256) void gemm_bf16_kernel(
    const unsigned short* __restrict__ A, const unsigned short* __restrict__ Bm,
    const float* __restrict__ bias, const float* __restrict__ res,
    void* __restrict__ Cp, int flags) {
  // double-buffered LDS tiles
  __shared__ __align__(16) unsigned short As[2][GBM][GBK];
  __shared__ __align__(16) unsigned short Bs[2][GBK][GBN];
  int tid = threadIdx.x;
  int lane = tid & 31, wave = tid >> 5;
  int wm = wave & 3, wn = wave >> 2;     // 4 x 2 wave grid -> 128 x 64 block tile
  int half = lane >> 4, lm = lane & 15;
  int bm = blockIdx.y * GBM, bn = blockIdx.x * GBN;

  // staging indices (computed once)
  int ar = tid >> 1;                 // 0..127 : A tile row
  int ac = (tid & 1) << 4;           // 0 or 16: A tile half-row
  int bkk = tid >> 3;                // 0..31  : B tile row
  int bnn = (tid & 7) << 3;          // 0..56  : B tile col (8 bf16 chunk)
  const unsigned short* aptr = A + (size_t)(bm + ar) * KDIM + ac;
  const unsigned short* bptr = Bm + (size_t)bkk * NDIM + bn + bnn;

  // this thread's LDS staging slots (byte addresses) for each buffer
  unsigned ldsA = lds_addr(&As[0][ar][ac]);
  unsigned ldsB = lds_addr(&Bs[0][bkk][bnn]);
  const unsigned strA = (unsigned)sizeof(As[0]);   // 8 KB
  const unsigned strB = (unsigned)sizeof(Bs[0]);   // 4 KB

  // prologue: async-stage K-tile 0 into buffer 0
  async_ld_b128(ldsA,      aptr);
  async_ld_b128(ldsA + 16, aptr + 8);
  async_ld_b128(ldsB,      bptr);
  aptr += GBK;
  bptr += (size_t)GBK * NDIM;

  v8f acc[2][2] = {};
  int buf = 0;
  for (int k0 = 0; k0 < KDIM; k0 += GBK) {
    wait_async0();                         // this wave's current tile landed in LDS
    __syncthreads();                       // all waves' tiles visible; frees buf^1
    if (k0 + GBK < KDIM) {                 // DMA next tile while we do the WMMAs
      unsigned ao = ldsA + (buf ^ 1) * strA;
      unsigned bo = ldsB + (buf ^ 1) * strB;
      async_ld_b128(ao,      aptr);
      async_ld_b128(ao + 16, aptr + 8);
      async_ld_b128(bo,      bptr);
      aptr += GBK;
      bptr += (size_t)GBK * NDIM;
      __builtin_prefetch(aptr, 0, 0);      // global_prefetch_b8 for the tile after
      __builtin_prefetch(bptr, 0, 0);
    }

    // fragments from current buffer
    Frag af[2], bfr[2];
#pragma unroll
    for (int t = 0; t < 2; ++t) {
      int row = wm * 32 + t * 16 + lm;
      int col = wn * 32 + t * 16 + lm;
#pragma unroll
      for (int vv = 0; vv < 8; ++vv) {
        af[t].u[vv] = *(const unsigned int*)&As[buf][row][pairK(vv, half)];
        int kk = pairK(vv, half);
        unsigned int lo = Bs[buf][kk][col];
        unsigned int hi = Bs[buf][kk + 1][col];
        bfr[t].u[vv] = lo | (hi << 16);
      }
    }
#pragma unroll
    for (int i = 0; i < 2; ++i)
#pragma unroll
      for (int j = 0; j < 2; ++j)
        acc[i][j] = wmma_bf16(af[i], bfr[j], acc[i][j]);

    buf ^= 1;
  }

  bool outbf = (flags & GF_OUT_BF16) != 0;
  bool relu  = (flags & GF_RELU) != 0;
#pragma unroll
  for (int i = 0; i < 2; ++i)
#pragma unroll
    for (int j = 0; j < 2; ++j) {
      int colg = bn + wn * 32 + j * 16 + lm;
      float bv = bias ? bias[colg] : 0.f;
#pragma unroll
      for (int r = 0; r < 8; ++r) {
        int rowg = bm + wm * 32 + i * 16 + r + 8 * half;
        float val = acc[i][j][r] + bv;
        if (relu) val = fmaxf(val, 0.f);
        size_t idx = (size_t)rowg * NDIM + colg;
        if (res) val += res[idx];
        if (outbf) ((unsigned short*)Cp)[idx] = f2bf(val);
        else       ((float*)Cp)[idx] = val;
      }
    }
}

// ---------------- flash attention: one wave per (b,h,16-query tile) ----------------
#define NEG_BIG (-3.0e38f)

__global__ __launch_bounds__(32) void attn_kernel(
    const unsigned short* __restrict__ Q, const unsigned short* __restrict__ Km,
    const unsigned short* __restrict__ V, unsigned short* __restrict__ O) {
  const int E = 1024, T = 2048, DH = 64;
  __shared__ __align__(16) unsigned short Ps[16][32];
  int lane = threadIdx.x & 31;
  int half = lane >> 4, lm = lane & 15;
  int qt = blockIdx.x;                 // 0..T/16-1
  int bh = blockIdx.y;                 // 0..B*H-1
  int b = bh >> 4, h = bh & 15;
  size_t rowbase = (size_t)b * T;
  int q0 = qt * 16;
  int hoff = h * DH;
  const float scale = 0.03125f;        // 1024^-0.5 (reference scales by full E)

  // Q A-fragments (rows q0+lm, d-chunks 0..31 / 32..63), loaded once
  Frag qa[2];
  const unsigned short* qrow = Q + (rowbase + q0 + lm) * E + hoff;
#pragma unroll
  for (int t = 0; t < 2; ++t)
#pragma unroll
    for (int vv = 0; vv < 8; ++vv)
      qa[t].u[vv] = *(const unsigned int*)(qrow + t * 32 + pairK(vv, half));

  float mrun[8], lrun[8];
#pragma unroll
  for (int r = 0; r < 8; ++r) { mrun[r] = NEG_BIG; lrun[r] = 0.f; }
  v8f oacc[4] = {};

  int qmax = q0 + 15;
  for (int kt = 0; kt <= qmax; kt += 32) {
    // scores: two 16-key subtiles x two 32-wide d-chunks; batch all loads first
    const unsigned short* krow0 = Km + (rowbase + kt + lm) * E + hoff;
    const unsigned short* krow1 = krow0 + (size_t)16 * E;
    Frag kf[4];
#pragma unroll
    for (int t = 0; t < 2; ++t)
#pragma unroll
      for (int vv = 0; vv < 8; ++vv) {
        kf[t].u[vv]     = *(const unsigned int*)(krow0 + t * 32 + pairK(vv, half));
        kf[2 + t].u[vv] = *(const unsigned int*)(krow1 + t * 32 + pairK(vv, half));
      }
    v8f s0 = {}, s1 = {};
    s0 = wmma_bf16(qa[0], kf[0], s0);
    s0 = wmma_bf16(qa[1], kf[1], s0);
    s1 = wmma_bf16(qa[0], kf[2], s1);
    s1 = wmma_bf16(qa[1], kf[3], s1);

    // online softmax update (row = r + 8*half, 16 lanes per row share a half)
#pragma unroll
    for (int r = 0; r < 8; ++r) {
      int mq = q0 + r + 8 * half;
      float a0 = s0[r] * scale; if (kt + lm > mq)      a0 = NEG_BIG;
      float a1 = s1[r] * scale; if (kt + 16 + lm > mq) a1 = NEG_BIG;
      float mx = fmaxf(a0, a1);
#pragma unroll
      for (int off = 1; off < 16; off <<= 1) mx = fmaxf(mx, __shfl_xor(mx, off, 16));
      float mn = fmaxf(mrun[r], mx);
      float alpha = __expf(mrun[r] - mn);
      float p0 = __expf(a0 - mn), p1 = __expf(a1 - mn);
      float rs = p0 + p1;
#pragma unroll
      for (int off = 1; off < 16; off <<= 1) rs += __shfl_xor(rs, off, 16);
      lrun[r] = lrun[r] * alpha + rs;
      mrun[r] = mn;
#pragma unroll
      for (int j = 0; j < 4; ++j) oacc[j][r] *= alpha;
      Ps[r + 8 * half][lm]      = f2bf(p0);
      Ps[r + 8 * half][lm + 16] = f2bf(p1);
    }
    __syncthreads();

    // P (16x32) as A-fragment from LDS; V (32x16 per d-subtile) gathered from global
    Frag pf;
#pragma unroll
    for (int vv = 0; vv < 8; ++vv)
      pf.u[vv] = *(const unsigned int*)&Ps[lm][pairK(vv, half)];
    const unsigned short* vbase = V + (rowbase + kt) * E + hoff + lm;
    Frag vf[4];
#pragma unroll
    for (int j = 0; j < 4; ++j) {
      const unsigned short* vc = vbase + j * 16;
#pragma unroll
      for (int vv = 0; vv < 8; ++vv) {
        int kk = pairK(vv, half);
        unsigned int lo = vc[(size_t)kk * E];
        unsigned int hi = vc[(size_t)(kk + 1) * E];
        vf[j].u[vv] = lo | (hi << 16);
      }
    }
#pragma unroll
    for (int j = 0; j < 4; ++j)
      oacc[j] = wmma_bf16(pf, vf[j], oacc[j]);
    __syncthreads();
  }

  // epilogue: O /= l, store bf16
#pragma unroll
  for (int r = 0; r < 8; ++r) {
    float inv = 1.0f / lrun[r];
    size_t rowg = rowbase + q0 + r + 8 * half;
#pragma unroll
    for (int j = 0; j < 4; ++j)
      O[rowg * E + hoff + j * 16 + lm] = f2bf(oacc[j][r] * inv);
  }
}

// ---------------- host launcher ----------------
extern "C" void kernel_launch(void* const* d_in, const int* in_sizes, int n_in,
                              void* d_out, int out_size, void* d_ws, size_t ws_size,
                              hipStream_t stream) {
  const int E = 1024, H = 16, DFF = 4096, B = 2, T = 2048;
  const int M = B * T;                 // 4096 tokens
  (void)in_sizes; (void)n_in; (void)out_size; (void)ws_size;

  const float* x   = (const float*)d_in[0];
  const float* Wq  = (const float*)d_in[1];
  const float* Wk  = (const float*)d_in[2];
  const float* Wv  = (const float*)d_in[3];
  const float* Wo  = (const float*)d_in[4];
  const float* bo  = (const float*)d_in[5];
  const float* W1  = (const float*)d_in[6];
  const float* b1  = (const float*)d_in[7];
  const float* W2  = (const float*)d_in[8];
  const float* b2  = (const float*)d_in[9];
  const float* g1  = (const float*)d_in[10];
  const float* be1 = (const float*)d_in[11];
  const float* g2  = (const float*)d_in[12];
  const float* be2 = (const float*)d_in[13];
  float* out = (float*)d_out;

  // workspace carve-up (256B aligned)
  char* wsp = (char*)d_ws;
  auto alloc = [&](size_t bytes) -> void* {
    void* p = (void*)wsp;
    wsp += (bytes + 255) & ~(size_t)255;
    return p;
  };
  const size_t EE = (size_t)E * E, ED = (size_t)E * DFF;
  unsigned short* wq16 = (unsigned short*)alloc(EE * 2);
  unsigned short* wk16 = (unsigned short*)alloc(EE * 2);
  unsigned short* wv16 = (unsigned short*)alloc(EE * 2);
  unsigned short* wo16 = (unsigned short*)alloc(EE * 2);
  unsigned short* w116 = (unsigned short*)alloc(ED * 2);
  unsigned short* w216 = (unsigned short*)alloc(ED * 2);
  unsigned short* hb   = (unsigned short*)alloc((size_t)M * E * 2);
  unsigned short* qb   = (unsigned short*)alloc((size_t)(M + 64) * E * 2);  // +64 row pad
  unsigned short* kb   = (unsigned short*)alloc((size_t)(M + 64) * E * 2);
  unsigned short* vb   = (unsigned short*)alloc((size_t)(M + 64) * E * 2);
  unsigned short* ob   = (unsigned short*)alloc((size_t)M * E * 2);
  float*          x1   = (float*)alloc((size_t)M * E * 4);
  unsigned short* h2   = (unsigned short*)alloc((size_t)M * E * 2);
  unsigned short* f1   = (unsigned short*)alloc((size_t)M * DFF * 2);

  // weights -> bf16
  cvt_bf16_kernel<<<(EE + 1023) / 1024, 256, 0, stream>>>(Wq, wq16, (int)EE);
  cvt_bf16_kernel<<<(EE + 1023) / 1024, 256, 0, stream>>>(Wk, wk16, (int)EE);
  cvt_bf16_kernel<<<(EE + 1023) / 1024, 256, 0, stream>>>(Wv, wv16, (int)EE);
  cvt_bf16_kernel<<<(EE + 1023) / 1024, 256, 0, stream>>>(Wo, wo16, (int)EE);
  cvt_bf16_kernel<<<(ED + 1023) / 1024, 256, 0, stream>>>(W1, w116, (int)ED);
  cvt_bf16_kernel<<<(ED + 1023) / 1024, 256, 0, stream>>>(W2, w216, (int)ED);

  // LN1
  ln_bf16_kernel<<<M, 256, 0, stream>>>(x, g1, be1, hb);

  // QKV projections
  dim3 gEE(E / GBN, M / GBM);
  gemm_bf16_kernel<1024, 1024><<<gEE, 256, 0, stream>>>(hb, wq16, nullptr, nullptr, qb, GF_OUT_BF16);
  gemm_bf16_kernel<1024, 1024><<<gEE, 256, 0, stream>>>(hb, wk16, nullptr, nullptr, kb, GF_OUT_BF16);
  gemm_bf16_kernel<1024, 1024><<<gEE, 256, 0, stream>>>(hb, wv16, nullptr, nullptr, vb, GF_OUT_BF16);

  // causal flash attention
  attn_kernel<<<dim3(T / 16, B * H), 32, 0, stream>>>(qb, kb, vb, ob);

  // out-proj + bias + residual(x) -> x1 (fp32)
  gemm_bf16_kernel<1024, 1024><<<gEE, 256, 0, stream>>>(ob, wo16, bo, x, x1, 0);

  // LN2
  ln_bf16_kernel<<<M, 256, 0, stream>>>(x1, g2, be2, h2);

  // FFN
  dim3 gFF1(DFF / GBN, M / GBM);
  gemm_bf16_kernel<4096, 1024><<<gFF1, 256, 0, stream>>>(h2, w116, b1, nullptr, f1,
                                                         GF_OUT_BF16 | GF_RELU);
  gemm_bf16_kernel<1024, 4096><<<gEE, 256, 0, stream>>>(f1, w216, b2, x1, out, 0);
}
